// CustomGPT2Attention_70824010711563
// MI455X (gfx1250) — matrix operand order repaired and checked by reading, not compiled, not run on hardware
//
#include <hip/hip_runtime.h>
#include <hip/hip_bf16.h>

// ---------------------------------------------------------------------------
// Problem constants (from reference): B=2, S=2048, D=1024, H=16, HD=64
// ---------------------------------------------------------------------------
constexpr int BSZ = 2;
constexpr int SEQ = 2048;
constexpr int DIM = 1024;
constexpr int NH  = 16;
constexpr int HD  = 64;
constexpr int NQKV = 3 * DIM;

typedef __attribute__((ext_vector_type(16))) __bf16 bf16x16;
typedef __attribute__((ext_vector_type(8)))  float  f32x8;

// Pointer types matching the async-LDS builtin signature: V4i*1 / V4i*3
typedef int v4i __attribute__((vector_size(16)));
typedef __attribute__((address_space(1))) v4i gv4i;
typedef __attribute__((address_space(3))) v4i lv4i;
typedef __attribute__((address_space(3))) const char as3_cchar;

// TDM descriptor group types (per CDNA5_HIP.md probe)
typedef unsigned int u32x4 __attribute__((ext_vector_type(4)));
typedef int          i32x4 __attribute__((ext_vector_type(4)));
typedef int          i32x8 __attribute__((ext_vector_type(8)));

// ---------------------------------------------------------------------------
// CDNA5 async global->LDS copy (ASYNCcnt path), with safe fallbacks.
// ---------------------------------------------------------------------------
#if __has_builtin(__builtin_amdgcn_global_load_async_to_lds_b128)
#define HAVE_ASYNC_LDS 1
#else
#define HAVE_ASYNC_LDS 0
#endif

#if __has_builtin(__builtin_amdgcn_tensor_load_to_lds)
#define HAVE_TDM 1
#else
#define HAVE_TDM 0
#endif

__device__ inline void async_copy16(const __bf16* g, __bf16* l) {
#if HAVE_ASYNC_LDS
  __builtin_amdgcn_global_load_async_to_lds_b128((gv4i*)g, (lv4i*)l,
                                                 /*offset=*/0, /*cpol=*/0);
#else
  *(uint4*)l = *(const uint4*)g;
#endif
}

__device__ inline void wait_async() {
#if HAVE_ASYNC_LDS
#if __has_builtin(__builtin_amdgcn_s_wait_asynccnt)
  __builtin_amdgcn_s_wait_asynccnt(0);
#else
  asm volatile("s_wait_asynccnt 0x0" ::: "memory");
#endif
#endif
}

__device__ inline unsigned lds_offset_of(const void* p) {
  return (unsigned)(size_t)(as3_cchar*)p;
}

#if HAVE_TDM
// 2D TDM load: tile (rows x 64) of bf16, row stride 64 elems in global,
// LDS rows padded to 72 elems (row = 32 DWORDs, pad 4 DWORDs).
// D# layout per CDNA5 ISA ch.8 (group0 128b, group1 256b).
__device__ inline void tdm_load_tile64(const __bf16* gsrc, unsigned lds_off,
                                       int rows) {
  const unsigned long long ga = (unsigned long long)(size_t)gsrc;
  u32x4 g0;
  g0[0] = 1u;                                        // count=1, user D#
  g0[1] = lds_off;                                   // lds_addr (bytes)
  g0[2] = (unsigned)(ga & 0xffffffffu);              // global_addr[31:0]
  g0[3] = (unsigned)((ga >> 32) & 0x01ffffffu)       // global_addr[56:32]
          | (2u << 30);                              // type=2 ("image")
  i32x8 g1;
  g1[0] = (1 << 16)      // data_size = 2 bytes
        | (1 << 20)      // pad_enable
        | (4 << 22)      // pad_interval: 32 DWORDs (one 64-elem bf16 row)
        | (3 << 25);     // pad_amount: 4 DWORDs (8 bf16 -> pitch 72)
  g1[1] = (64 & 0xffff) << 16;        // tensor_dim0[15:0] = 64
  g1[2] = (rows & 0xffff) << 16;      // tensor_dim0 hi=0 | tensor_dim1[15:0]
  g1[3] = (64 << 16);                 // tensor_dim1 hi=0 | tile_dim0 = 64
  g1[4] = rows & 0xffff;              // tile_dim1 = rows, tile_dim2 = 0
  g1[5] = 64;                         // tensor_dim0_stride[31:0] = 64 elems
  g1[6] = 0;                          // stride hi | tensor_dim1_stride lo
  g1[7] = 0;
  const i32x4 z4 = {0, 0, 0, 0};
#if defined(__clang_major__) && __clang_major__ >= 23
  const i32x8 z8 = {0, 0, 0, 0, 0, 0, 0, 0};
  __builtin_amdgcn_tensor_load_to_lds(g0, g1, z4, z4, z8, 0);
#else
  __builtin_amdgcn_tensor_load_to_lds(g0, g1, z4, z4, 0);
#endif
}

__device__ inline void wait_tensor() {
#if __has_builtin(__builtin_amdgcn_s_wait_tensorcnt)
  __builtin_amdgcn_s_wait_tensorcnt(0);
#else
  asm volatile("s_wait_tensorcnt 0x0" ::: "memory");
#endif
}
#endif  // HAVE_TDM

// ---------------------------------------------------------------------------
// WMMA wrapper: D = A(16x32 bf16) * B(32x16 bf16) + C(16x16 f32)
// ---------------------------------------------------------------------------
__device__ inline f32x8 wmma_bf16(bf16x16 a, bf16x16 b, f32x8 c) {
  return __builtin_amdgcn_wmma_f32_16x16x32_bf16(
      /*neg_a=*/false, a, /*neg_b=*/false, b,
      /*c_mod=*/(short)0, c, /*reuse_a=*/false, /*reuse_b=*/false);
}

// A fragment (16x32, row-major LDS source):
//   lane L (m = L&15): elem e -> K=(L>>4)*16+e
// With 16B-aligned rows each half is one contiguous 16B chunk (ds_load_b128-able).
__device__ inline bf16x16 load_a(const __bf16* s, int ld, int row0, int col0) {
  const int lane = threadIdx.x & 31;
  const int m  = row0 + (lane & 15);
  const int kg = (lane >> 4) * 16;
  bf16x16 a;
#pragma unroll
  for (int e = 0; e < 8; ++e) a[e]     = s[m * ld + col0 + kg + e];
#pragma unroll
  for (int e = 0; e < 8; ++e) a[e + 8] = s[m * ld + col0 + kg + 8 + e];
  return a;
}

// B fragment (32x16, source row-major [k][n]): lane L (n=L&15), elem e -> K=(L>>4)*16+e
__device__ inline bf16x16 load_b(const __bf16* s, int ld, int krow0, int ncol0) {
  const int lane = threadIdx.x & 31;
  const int n  = ncol0 + (lane & 15);
  const int kh = (lane >> 4) * 16;
  bf16x16 b;
#pragma unroll
  for (int e = 0; e < 16; ++e) b[e] = s[(krow0 + kh + e) * ld + n];
  return b;
}

// B fragment from a TRANSPOSED source: logical B[k][n] read from s[n][k].
__device__ inline bf16x16 load_bT(const __bf16* s, int ld, int krow0, int ncol0) {
  const int lane = threadIdx.x & 31;
  const int n  = ncol0 + (lane & 15);
  const int kh = (lane >> 4) * 16;
  bf16x16 b;
#pragma unroll
  for (int e = 0; e < 16; ++e) b[e] = s[n * ld + krow0 + kh + e];
  return b;
}

// ---------------------------------------------------------------------------
// Kernel 1: QKV = X @ W_qkv + b, RoPE on Q/K, scatter to bf16 [B,H,S,HD].
// Block: 128 threads (4 waves). Tile: 64(M) x 64(N), BK=32.
// ---------------------------------------------------------------------------
__global__ __launch_bounds__(128) void qkv_rope_kernel(
    const float* __restrict__ X, const float* __restrict__ W,
    const float* __restrict__ bias,
    __bf16* __restrict__ qb, __bf16* __restrict__ kb, __bf16* __restrict__ vb) {
  __shared__ alignas(16) __bf16 As[64][40];  // 80B rows (16B aligned)
  __shared__ alignas(16) __bf16 Bs[32][66];
  __shared__ alignas(16) float  Cs[64][64];

  const int tid  = threadIdx.x;
  const int lane = tid & 31;
  const int wave = tid >> 5;
  const int row0 = blockIdx.y * 64;  // M offset into B*S
  const int col0 = blockIdx.x * 64;  // N offset into 3*D

  f32x8 acc[4];
#pragma unroll
  for (int i = 0; i < 4; ++i)
#pragma unroll
    for (int r = 0; r < 8; ++r) acc[i][r] = 0.0f;

  for (int k0 = 0; k0 < DIM; k0 += 32) {
    {  // stage A tile 64x32 (f32 -> bf16)
      const int r = tid >> 1, c = (tid & 1) * 16;
      const float* src = X + (size_t)(row0 + r) * DIM + k0 + c;
#pragma unroll
      for (int j = 0; j < 16; j += 4) {
        float4 f = *(const float4*)(src + j);
        As[r][c + j + 0] = (__bf16)f.x; As[r][c + j + 1] = (__bf16)f.y;
        As[r][c + j + 2] = (__bf16)f.z; As[r][c + j + 3] = (__bf16)f.w;
      }
      if (k0 + 32 < DIM) __builtin_prefetch(src + 32, 0, 1);
    }
    {  // stage B tile 32x64
      const int r = tid >> 2, c = (tid & 3) * 16;
      const float* src = W + (size_t)(k0 + r) * NQKV + col0 + c;
#pragma unroll
      for (int j = 0; j < 16; j += 4) {
        float4 f = *(const float4*)(src + j);
        Bs[r][c + j + 0] = (__bf16)f.x; Bs[r][c + j + 1] = (__bf16)f.y;
        Bs[r][c + j + 2] = (__bf16)f.z; Bs[r][c + j + 3] = (__bf16)f.w;
      }
      if (k0 + 32 < DIM) __builtin_prefetch(src + (size_t)32 * NQKV, 0, 1);
    }
    __syncthreads();
    bf16x16 bfrag = load_b(&Bs[0][0], 66, 0, wave * 16);
#pragma unroll
    for (int mb = 0; mb < 4; ++mb) {
      bf16x16 afrag = load_a(&As[0][0], 40, mb * 16, 0);
      acc[mb] = wmma_bf16(afrag, bfrag, acc[mb]);
    }
    __syncthreads();
  }

  {  // spill accumulators to LDS f32 tile (C layout: row = (lane>>4)*8 + r)
    const int n = lane & 15, mb2 = (lane >> 4) * 8;
#pragma unroll
    for (int mb = 0; mb < 4; ++mb)
#pragma unroll
      for (int r = 0; r < 8; ++r)
        Cs[mb * 16 + mb2 + r][wave * 16 + n] = acc[mb][r];
  }
  __syncthreads();

  // Epilogue: bias + RoPE + scatter to bf16 [B,H,S,HD]
  for (int i = tid; i < 64 * 64; i += 128) {
    const int r = i >> 6, c = i & 63;
    const int nglob = col0 + c;
    const int mglob = row0 + r;
    const int mat = nglob >> 10;            // 0=q 1=k 2=v
    const int h   = (nglob & 1023) >> 6;
    const int d   = c;                      // col tile is 64-aligned
    const int b   = mglob >> 11;
    const int sp  = mglob & 2047;
    const float val = Cs[r][c] + bias[nglob];
    float outv;
    __bf16* dst;
    if (mat == 2) {
      outv = val;
      dst  = vb;
    } else {
      const int f = d & 31;
      // inv_freq = 10000^(-f/32) = exp(-f * ln(10000)/32)
      const float ang = (float)sp * __expf(-(float)f * 0.28782313663f);
      float sn, cn;
      __sincosf(ang, &sn, &cn);
      const int c2 = (d < 32) ? c + 32 : c - 32;
      const float val2 = Cs[r][c2] + bias[col0 + c2];
      outv = val * cn + ((d < 32) ? -val2 : val2) * sn;
      dst  = (mat == 0) ? qb : kb;
    }
    dst[(((size_t)b * NH + h) * SEQ + sp) * HD + d] = (__bf16)outv;
  }
}

// ---------------------------------------------------------------------------
// Kernel 2: causal flash attention per (b,h). Block: 256 threads (8 waves),
// q-tile = 128 rows (16 per wave), k-step = 32 keys. Q staged via TDM
// (tensor_load_to_lds); K/V staged via async global->LDS B128 copies.
// ---------------------------------------------------------------------------
__global__ __launch_bounds__(256) void attn_kernel(
    const __bf16* __restrict__ qb, const __bf16* __restrict__ kb,
    const __bf16* __restrict__ vb, __bf16* __restrict__ attnb) {
  __shared__ alignas(16) __bf16 Qs[128][72];  // 144B rows (16B aligned)
  __shared__ alignas(16) __bf16 Ks[32][72];
  __shared__ alignas(16) __bf16 Vs[32][72];
  __shared__ alignas(16) __bf16 Ps[8][16][40];

  const int tid  = threadIdx.x;
  const int lane = tid & 31;
  const int wave = tid >> 5;
  const int qbase = blockIdx.x * 128;
  const int bh    = blockIdx.y;
  const __bf16* Qp = qb + (size_t)bh * SEQ * HD;
  const __bf16* Kp = kb + (size_t)bh * SEQ * HD;
  const __bf16* Vp = vb + (size_t)bh * SEQ * HD;

#if HAVE_TDM
  // stage Q tile 128x64 bf16 with one TDM descriptor (wave 0 issues)
  if (wave == 0) {
    tdm_load_tile64(Qp + (size_t)qbase * HD, lds_offset_of(&Qs[0][0]), 128);
    wait_tensor();
  }
#else
  for (int i = tid; i < 128 * 8; i += 256) {
    const int r = i >> 3, c = (i & 7) * 8;
    async_copy16(Qp + (size_t)(qbase + r) * HD + c, &Qs[r][c]);
  }
  wait_async();
#endif
  __syncthreads();

  bf16x16 qf[2];
  qf[0] = load_a(&Qs[0][0], 72, wave * 16, 0);
  qf[1] = load_a(&Qs[0][0], 72, wave * 16, 32);

  f32x8 ao[4];
#pragma unroll
  for (int i = 0; i < 4; ++i)
#pragma unroll
    for (int r = 0; r < 8; ++r) ao[i][r] = 0.0f;
  float mrow[8], lrow[8];
#pragma unroll
  for (int r = 0; r < 8; ++r) { mrow[r] = -3.0e38f; lrow[r] = 0.0f; }

  const int kend = qbase + 128;
  for (int kb2 = 0; kb2 < kend; kb2 += 32) {
    __syncthreads();  // previous iteration's Ks/Vs/Ps reads complete
    for (int i = tid; i < 32 * 8; i += 256) {  // 256 chunks -> 1 per thread
      const int r = i >> 3, c = (i & 7) * 8;
      async_copy16(Kp + (size_t)(kb2 + r) * HD + c, &Ks[r][c]);
      async_copy16(Vp + (size_t)(kb2 + r) * HD + c, &Vs[r][c]);
    }
    wait_async();
    __syncthreads();

    // scores: S(16q x 32k) = Q(16x64) * K^T ; B read transposed from Ks[key][d]
    f32x8 sc[2];
#pragma unroll
    for (int nk = 0; nk < 2; ++nk) {
#pragma unroll
      for (int r = 0; r < 8; ++r) sc[nk][r] = 0.0f;
#pragma unroll
      for (int dc = 0; dc < 2; ++dc) {
        bf16x16 kfrag = load_bT(&Ks[0][0], 72, dc * 32, nk * 16);
        sc[nk] = wmma_bf16(qf[dc], kfrag, sc[nk]);
      }
    }

    // scale + causal mask
#pragma unroll
    for (int nk = 0; nk < 2; ++nk)
#pragma unroll
      for (int r = 0; r < 8; ++r) {
        const float v = sc[nk][r] * 0.125f;  // 1/sqrt(64)
        const int kc = kb2 + nk * 16 + (lane & 15);
        const int qi = qbase + wave * 16 + (lane >> 4) * 8 + r;
        sc[nk][r] = (kc <= qi) ? v : -3.0e38f;
      }

    // online softmax: each row lives in a 16-lane half; reduce via shfl_xor
#pragma unroll
    for (int r = 0; r < 8; ++r) {
      float mx = fmaxf(sc[0][r], sc[1][r]);
#pragma unroll
      for (int off = 8; off >= 1; off >>= 1)
        mx = fmaxf(mx, __shfl_xor(mx, off, 32));
      const float mnew  = fmaxf(mrow[r], mx);
      const float alpha = __expf(mrow[r] - mnew);
      const float p0 = __expf(sc[0][r] - mnew);
      const float p1 = __expf(sc[1][r] - mnew);
      float rs = p0 + p1;
#pragma unroll
      for (int off = 8; off >= 1; off >>= 1) rs += __shfl_xor(rs, off, 32);
      lrow[r] = lrow[r] * alpha + rs;
      mrow[r] = mnew;
      sc[0][r] = p0; sc[1][r] = p1;
#pragma unroll
      for (int dc = 0; dc < 4; ++dc) ao[dc][r] *= alpha;
    }

    // re-lay P (C layout) into A-fragment form through LDS
    {
      const int n = lane & 15, mb = (lane >> 4) * 8;
#pragma unroll
      for (int nk = 0; nk < 2; ++nk)
#pragma unroll
        for (int r = 0; r < 8; ++r)
          Ps[wave][mb + r][nk * 16 + n] = (__bf16)sc[nk][r];
    }
    __syncthreads();
    bf16x16 pf = load_a(&Ps[wave][0][0], 40, 0, 0);
#pragma unroll
    for (int dc = 0; dc < 4; ++dc) {
      bf16x16 vfrag = load_b(&Vs[0][0], 72, 0, dc * 16);
      ao[dc] = wmma_bf16(pf, vfrag, ao[dc]);
    }
  }

  // normalize and store (bf16) directly in (B,S,D) layout
  {
    const int n = lane & 15, mb = (lane >> 4) * 8;
    const int b = bh >> 4, h = bh & 15;
#pragma unroll
    for (int r = 0; r < 8; ++r) {
      const float inv = 1.0f / lrow[r];
      const int srow  = qbase + wave * 16 + mb + r;
      __bf16* dst = attnb + ((size_t)b * SEQ + srow) * DIM + h * HD;
#pragma unroll
      for (int dc = 0; dc < 4; ++dc)
        dst[dc * 16 + n] = (__bf16)(ao[dc][r] * inv);
    }
  }
}

// ---------------------------------------------------------------------------
// Kernel 3: out = attn @ W_proj + b_proj. A tile staged via async B128
// (attn scratch already bf16); W tile converted f32->bf16.
// ---------------------------------------------------------------------------
__global__ __launch_bounds__(128) void proj_kernel(
    const __bf16* __restrict__ Xa, const float* __restrict__ W,
    const float* __restrict__ bias, float* __restrict__ out) {
  __shared__ alignas(16) __bf16 As[64][40];  // 80B rows (16B aligned)
  __shared__ alignas(16) __bf16 Bs[32][66];

  const int tid  = threadIdx.x;
  const int lane = tid & 31;
  const int wave = tid >> 5;
  const int row0 = blockIdx.y * 64;
  const int col0 = blockIdx.x * 64;

  f32x8 acc[4];
#pragma unroll
  for (int i = 0; i < 4; ++i)
#pragma unroll
    for (int r = 0; r < 8; ++r) acc[i][r] = 0.0f;

  for (int k0 = 0; k0 < DIM; k0 += 32) {
    // stage A tile 64x32 bf16 via async copies: 4 chunks/row, 256 total
    for (int i = tid; i < 64 * 4; i += 128) {
      const int r = i >> 2, c = (i & 3) * 8;
      async_copy16(Xa + (size_t)(row0 + r) * DIM + k0 + c, &As[r][c]);
    }
    {  // stage B tile 32x64 (f32 -> bf16)
      const int r = tid >> 2, c = (tid & 3) * 16;
      const float* src = W + (size_t)(k0 + r) * DIM + col0 + c;
#pragma unroll
      for (int j = 0; j < 16; j += 4) {
        float4 f = *(const float4*)(src + j);
        Bs[r][c + j + 0] = (__bf16)f.x; Bs[r][c + j + 1] = (__bf16)f.y;
        Bs[r][c + j + 2] = (__bf16)f.z; Bs[r][c + j + 3] = (__bf16)f.w;
      }
      if (k0 + 32 < DIM) __builtin_prefetch(src + (size_t)32 * DIM, 0, 1);
    }
    wait_async();
    __syncthreads();
    bf16x16 bfrag = load_b(&Bs[0][0], 66, 0, wave * 16);
#pragma unroll
    for (int mb = 0; mb < 4; ++mb) {
      bf16x16 afrag = load_a(&As[0][0], 40, mb * 16, 0);
      acc[mb] = wmma_bf16(afrag, bfrag, acc[mb]);
    }
    __syncthreads();
  }

  const int n = lane & 15, mb2 = (lane >> 4) * 8;
#pragma unroll
  for (int mb = 0; mb < 4; ++mb)
#pragma unroll
    for (int r = 0; r < 8; ++r) {
      const int nglob = col0 + wave * 16 + n;
      out[(size_t)(row0 + mb * 16 + mb2 + r) * DIM + nglob] =
          acc[mb][r] + bias[nglob];
    }
}

// ---------------------------------------------------------------------------
// Host-side launcher
// ---------------------------------------------------------------------------
extern "C" void kernel_launch(void* const* d_in, const int* in_sizes, int n_in,
                              void* d_out, int out_size, void* d_ws, size_t ws_size,
                              hipStream_t stream) {
  (void)in_sizes; (void)n_in; (void)out_size; (void)ws_size;
  const float* X     = (const float*)d_in[0];
  const float* Wqkv  = (const float*)d_in[1];
  const float* bqkv  = (const float*)d_in[2];
  const float* Wproj = (const float*)d_in[3];
  const float* bproj = (const float*)d_in[4];
  float* out = (float*)d_out;

  const size_t planeN = (size_t)BSZ * SEQ * DIM;  // B*H*S*HD == B*S*D
  __bf16* qb    = (__bf16*)d_ws;
  __bf16* kbuf  = qb + planeN;
  __bf16* vb    = kbuf + planeN;
  __bf16* attnb = vb + planeN;

  dim3 g1(NQKV / 64, BSZ * SEQ / 64);  // (48, 64)
  qkv_rope_kernel<<<g1, 128, 0, stream>>>(X, Wqkv, bqkv, qb, kbuf, vb);

  dim3 g2(SEQ / 128, BSZ * NH);        // (16, 32)
  attn_kernel<<<g2, 256, 0, stream>>>(qb, kbuf, vb, attnb);

  dim3 g3(DIM / 64, BSZ * SEQ / 64);   // (16, 64)
  proj_kernel<<<g3, 128, 0, stream>>>(attnb, Wproj, bproj, out);
}
